// HybridGridQueryGenerator_13271448945234
// MI455X (gfx1250) — matrix operand order repaired and checked by reading, not compiled, not run on hardware
//
#include <hip/hip_runtime.h>
#include <hip/hip_bf16.h>
#include <math.h>

// Problem constants (match reference)
#define BB    64
#define MAXH  64
#define MAXW  32
#define DD    512
#define SS    8
#define HID   64
#define QOFF  ((size_t)BB * MAXH * MAXW * DD)   // offset of mask output in d_out

typedef __attribute__((ext_vector_type(16))) _Float16 v16h;
typedef __attribute__((ext_vector_type(8)))  _Float16 v8h;
typedef __attribute__((ext_vector_type(8)))  float    v8f;

// -------------------------------------------------------------------------
// Pack w2 (f32 [64][512], K x N row-major) into f16 WMMA-B fragment layout:
//   w2p[((ntile*2 + kstep)*32 + lane)*16 + j]
//     = w2[(kstep*32 + (lane&16 ? 16 : 0) + j) * 512 + ntile*16 + (lane&15)]
// B (32x16) k order used here, same as A: lanes 0-15 hold N=lane, K=0..15 across
// 8 VGPRs (halfs j=0..15 <-> K=kbase+j); lanes 16-31 hold K=16..31.
// -------------------------------------------------------------------------
__global__ __launch_bounds__(256) void hgq_pack_w2(const float* __restrict__ w2,
                                                   _Float16* __restrict__ w2p) {
    int t = blockIdx.x * 256 + threadIdx.x;              // 0..32767
    int j     =  t        & 15;
    int lane  = (t >> 4)  & 31;
    int kstep = (t >> 9)  &  1;
    int ntile =  t >> 10;
    int col   = ntile * 16 + (lane & 15);
    int k     = kstep * 32 + ((lane & 16) ? 16 : 0) + j;
    w2p[t] = (_Float16)w2[k * DD + col];
}

// -------------------------------------------------------------------------
// Main kernel: one workgroup = one batch b, two h-rows (64 locations),
// all 512 channels.  8 waves: wave w -> Mtile = w&3 (16 locations),
// Nhalf = w>>2 (16 of the 32 N-tiles).
// -------------------------------------------------------------------------
__global__ __launch_bounds__(256) void hgq_main(
        const float* __restrict__ canonical,   // [8][8][512]
        const float* __restrict__ w1,          // [2][64]
        const float* __restrict__ b1,          // [64]
        const float* __restrict__ b2,          // [512]
        const int*   __restrict__ th,          // [64]
        const int*   __restrict__ tw,          // [64]
        const _Float16* __restrict__ w2p,      // packed B fragments (64 KB)
        float* __restrict__ out)               // queries ++ mask
{
    __shared__ _Float16 sH [64 * 66];        // hidden, f16, stride 66 (8.25 KB)
    __shared__ _Float16 sYB[2 * SS * DD];    // wy-preblended rows, f16 (16 KB)
    __shared__ float    sB2[DD];             // bias (2 KB)
    __shared__ float    sT [8 * 256];        // per-wave transpose tile (8 KB)
    __shared__ float    sWX[64], sMK[64];
    __shared__ int      sX0[64];

    const int tid = threadIdx.x;
    const int b   = blockIdx.x >> 5;       // batch
    const int mt  = blockIdx.x & 31;       // which pair of h-rows
    const int h0  = mt * 2;

    const int Hs = min(max(th[b], 1), MAXH);
    const int Ws = min(max(tw[b], 1), MAXW);

    // Per-h interpolation setup for the two rows this WG owns
    float u0 = (Hs > 1) ? (float)(h0)     / (float)(Hs - 1) : 0.f;
    float u1 = (Hs > 1) ? (float)(h0 + 1) / (float)(Hs - 1) : 0.f;
    float sy0 = fminf(fmaxf(u0 * (SS - 1), 0.f), (float)(SS - 1));
    float sy1 = fminf(fmaxf(u1 * (SS - 1), 0.f), (float)(SS - 1));
    int   y00 = (int)floorf(sy0), y10 = (int)floorf(sy1);
    float wyA = sy0 - (float)y00, wyB = sy1 - (float)y10;
    int rows[4] = { y00, min(y00 + 1, SS - 1), y10, min(y10 + 1, SS - 1) };

    // ---- Phase 1a: stage canonical rows, pre-blending in y (f32 -> f16) ----
    // yb[hl][x][d] = (1-wy)*canon[y0][x][d] + wy*canon[y1][x][d]
    const float4* can4 = reinterpret_cast<const float4*>(canonical);
    for (int j = tid; j < 2048; j += 256) {          // 2 slots * 1024 float4
        int slot = j >> 10;                          // which h-row
        int rem  = j & 1023;                         // float4 index in row
        float4 r0 = can4[rows[2 * slot]     * (SS * DD / 4) + rem];
        float4 r1 = can4[rows[2 * slot + 1] * (SS * DD / 4) + rem];
        float wy  = slot ? wyB : wyA;
        float wm  = 1.f - wy;
        int o = slot * (SS * DD) + rem * 4;
        sYB[o + 0] = (_Float16)(wm * r0.x + wy * r1.x);
        sYB[o + 1] = (_Float16)(wm * r0.y + wy * r1.y);
        sYB[o + 2] = (_Float16)(wm * r0.z + wy * r1.z);
        sYB[o + 3] = (_Float16)(wm * r0.w + wy * r1.w);
    }
    // ---- Phase 1b: bias to LDS ----
    sB2[tid]       = b2[tid];
    sB2[tid + 256] = b2[tid + 256];

    // ---- Phase 1c: hidden = gelu([u,v] @ w1 + b1), f16 into LDS ----
    {
        int i  = tid >> 2;                 // location 0..63
        int kq = (tid & 3) << 4;           // 16 hidden units per thread
        int hl = i >> 5;                   // 0 or 1
        int wl = i & 31;
        float u = hl ? u1 : u0;
        float v = (Ws > 1) ? (float)wl / (float)(Ws - 1) : 0.f;
        #pragma unroll
        for (int k = kq; k < kq + 16; ++k) {
            float pre = u * w1[k] + v * w1[HID + k] + b1[k];
            float g = 0.5f * pre * (1.f + erff(pre * 0.70710678118654752f));
            sH[i * 66 + k] = (_Float16)g;
        }
        if ((tid & 3) == 0) {
            float sx = fminf(fmaxf(v * (SS - 1), 0.f), (float)(SS - 1));
            int   x0 = (int)floorf(sx);
            sX0[i] = x0;
            sWX[i] = sx - (float)x0;
            int hh = h0 + hl;
            float mkf = ((hh < Hs) && (wl < Ws)) ? 1.f : 0.f;
            sMK[i] = mkf;
            out[QOFF + (size_t)b * (MAXH * MAXW) + (size_t)mt * 64 + i] = mkf;
        }
    }
    __syncthreads();

    // ---- Phase 2: WMMA GEMM (64x64 @ 64x512) + vectorized epilogue ----
    const int wave   = tid >> 5;
    const int lane   = tid & 31;
    const int Mtile  = wave & 3;
    const int Nhalf  = wave >> 2;
    const int laneHi = (lane >> 4) & 1;
    const int nlo    = lane & 15;

    // A fragments (16-bit 16x32 layout), kstep 0 and 1
    v16h a0, a1;
    {
        const int mrow = Mtile * 16 + nlo;
        unsigned* a0u = reinterpret_cast<unsigned*>(&a0);
        unsigned* a1u = reinterpret_cast<unsigned*>(&a1);
        #pragma unroll
        for (int v = 0; v < 8; ++v) {
            int K0 = laneHi * 16 + 2 * v;
            a0u[v] = *reinterpret_cast<const unsigned*>(&sH[mrow * 66 + K0]);
            a1u[v] = *reinterpret_cast<const unsigned*>(&sH[mrow * 66 + 32 + K0]);
        }
    }

    // Store-side lane mapping: lane -> (row mr = lane>>1, 8 consecutive d)
    float* sTw = &sT[wave * 256];
    const int   mr   = lane >> 1;
    const int   dh   = (lane & 1) * 8;          // d sub-offset within tile
    const int   iloc = Mtile * 16 + mr;
    const int   x0   = sX0[iloc];
    const int   x1   = min(x0 + 1, SS - 1);
    const float wx   = sWX[iloc];
    const float wxm  = 1.f - wx;
    const float mkf  = sMK[iloc];
    const int   ybs  = (iloc >> 5) * (SS * DD); // which pre-blended h-row
    const int   hh   = h0 + (iloc >> 5);
    const int   ww   = iloc & 31;
    const size_t obase = ((size_t)((b * MAXH + hh) * MAXW + ww)) * DD + dh;

    for (int nt = 0; nt < 16; ++nt) {
        const int ntile = Nhalf * 16 + nt;
        const v16h bf0 = *reinterpret_cast<const v16h*>(
            w2p + ((size_t)(ntile * 2 + 0) * 32 + lane) * 16);
        const v16h bf1 = *reinterpret_cast<const v16h*>(
            w2p + ((size_t)(ntile * 2 + 1) * 32 + lane) * 16);
        if (nt + 1 < 16)   // keep next tile's fragments warm in L2/WGP$
            __builtin_prefetch(w2p + ((size_t)(ntile + 1) * 2 * 32 + lane) * 16, 0, 3);

        v8f acc = {};
        acc = __builtin_amdgcn_wmma_f32_16x16x32_f16(
                  false, a0, false, bf0, (short)0, acc, false, false);
        acc = __builtin_amdgcn_wmma_f32_16x16x32_f16(
                  false, a1, false, bf1, (short)0, acc, false, false);

        // Transpose acc tile through per-wave LDS scratch for b128 stores
        #pragma unroll
        for (int r = 0; r < 8; ++r)
            sTw[(r + laneHi * 8) * 16 + nlo] = acc[r];
        asm volatile("s_wait_dscnt 0x0" ::: "memory");  // per-wave WAR/RAW order

        float4 t0 = *reinterpret_cast<const float4*>(&sTw[mr * 16 + dh]);
        float4 t1 = *reinterpret_cast<const float4*>(&sTw[mr * 16 + dh + 4]);

        const int d0 = ntile * 16 + dh;
        float4 bias0 = *reinterpret_cast<const float4*>(&sB2[d0]);
        float4 bias1 = *reinterpret_cast<const float4*>(&sB2[d0 + 4]);
        v8h y0v = *reinterpret_cast<const v8h*>(&sYB[ybs + x0 * DD + d0]);
        v8h y1v = *reinterpret_cast<const v8h*>(&sYB[ybs + x1 * DD + d0]);

        float4 o0, o1;
        o0.x = mkf * (t0.x + bias0.x + (wxm * (float)y0v[0] + wx * (float)y1v[0]));
        o0.y = mkf * (t0.y + bias0.y + (wxm * (float)y0v[1] + wx * (float)y1v[1]));
        o0.z = mkf * (t0.z + bias0.z + (wxm * (float)y0v[2] + wx * (float)y1v[2]));
        o0.w = mkf * (t0.w + bias0.w + (wxm * (float)y0v[3] + wx * (float)y1v[3]));
        o1.x = mkf * (t1.x + bias1.x + (wxm * (float)y0v[4] + wx * (float)y1v[4]));
        o1.y = mkf * (t1.y + bias1.y + (wxm * (float)y0v[5] + wx * (float)y1v[5]));
        o1.z = mkf * (t1.z + bias1.z + (wxm * (float)y0v[6] + wx * (float)y1v[6]));
        o1.w = mkf * (t1.w + bias1.w + (wxm * (float)y0v[7] + wx * (float)y1v[7]));

        float* op = out + obase + (size_t)ntile * 16;
        *reinterpret_cast<float4*>(op)     = o0;
        *reinterpret_cast<float4*>(op + 4) = o1;
        // Wait for LDS reads of sTw before next iteration overwrites it
        // (per-wave LDS is in-order, but keep the compiler honest)
        asm volatile("" ::: "memory");
    }
}

extern "C" void kernel_launch(void* const* d_in, const int* in_sizes, int n_in,
                              void* d_out, int out_size, void* d_ws, size_t ws_size,
                              hipStream_t stream) {
    const float* canonical = (const float*)d_in[0];
    const float* w1        = (const float*)d_in[1];
    const float* b1        = (const float*)d_in[2];
    const float* w2        = (const float*)d_in[3];
    const float* b2        = (const float*)d_in[4];
    const int*   th        = (const int*)  d_in[6];
    const int*   tw        = (const int*)  d_in[7];

    _Float16* w2p = (_Float16*)d_ws;        // 64 KB packed B fragments
    float*    out = (float*)d_out;

    hgq_pack_w2<<<128, 256, 0, stream>>>(w2, w2p);
    hgq_main<<<BB * (MAXH / 2), 256, 0, stream>>>(
        canonical, w1, b1, b2, th, tw, w2p, out);
}
